// BMM_S8T_S8N_F32T_24859270709254
// MI455X (gfx1250) — 8-bit matrix operand order repaired and checked by reading, compiled, not run on hardware
//
#include <hip/hip_runtime.h>
#include <stdint.h>

typedef __attribute__((ext_vector_type(8))) int v8i;
typedef __attribute__((ext_vector_type(4))) int v4i;
typedef __attribute__((ext_vector_type(2))) int v2i;

namespace {
constexpr int kB = 256;
constexpr int kM = 512;
constexpr int kN = 512;
constexpr int kK = 128;
constexpr int kNTiles = 4;
constexpr int kWavesPerBlock = 8;
}

__global__ __launch_bounds__(256, 1)
void bmm_s8t_s8n_f32t_wmma(const int8_t* __restrict__ A,
                           const int8_t* __restrict__ Bmat,
                           const float* __restrict__ alpha_p,
                           float* __restrict__ Out)
{
    const int lane = threadIdx.x & 31;
    const int wave = threadIdx.x >> 5;
    const int gw   = blockIdx.x * kWavesPerBlock + wave;

    const int batch  = gw >> 8;
    const int rem    = gw & 255;
    const int m_base = (rem >> 3) << 4;
    const int n_base = (rem & 7) << 6;

    const int hi = lane >> 4;
    const int lm = lane & 15;

    const int8_t* a_row = A + ((size_t)batch * kM + (size_t)(m_base + lm)) * kK;
    v8i areg[2];
#pragma unroll
    for (int kb = 0; kb < 2; ++kb) {
        const int8_t* p = a_row + kb * 64 + hi * 16;
        v2i t0 = *(const v2i*)(p +  0);
        v2i t1 = *(const v2i*)(p +  8);
        v2i t2 = *(const v2i*)(p + 32);
        v2i t3 = *(const v2i*)(p + 40);
        v8i a;
        a[0] = t0[0]; a[1] = t0[1];
        a[2] = t1[0]; a[3] = t1[1];
        a[4] = t2[0]; a[5] = t2[1];
        a[6] = t3[0]; a[7] = t3[1];
        areg[kb] = a;
    }

    v8i breg[kNTiles][2];
#pragma unroll
    for (int nt = 0; nt < kNTiles; ++nt) {
        const int8_t* b_row =
            Bmat + ((size_t)batch * kN + (size_t)(n_base + nt * 16 + lm)) * kK;
#pragma unroll
        for (int kb = 0; kb < 2; ++kb) {
            const int8_t* p = b_row + kb * 64 + hi * 16;
            v4i lo  = *(const v4i*)(p +  0);
            v4i hi4 = *(const v4i*)(p + 32);
            v8i bb;
            bb[0] = lo[0];  bb[1] = lo[1];  bb[2] = lo[2];  bb[3] = lo[3];
            bb[4] = hi4[0]; bb[5] = hi4[1]; bb[6] = hi4[2]; bb[7] = hi4[3];
            breg[nt][kb] = bb;
        }
    }

    const float alpha = *alpha_p;

    float* out_base =
        Out + ((size_t)batch * kM + (size_t)(m_base + hi * 8)) * kN + n_base + lm;

#pragma unroll
    for (int nt = 0; nt < kNTiles; ++nt) {
        v8i acc = {};
        acc = __builtin_amdgcn_wmma_i32_16x16x64_iu8(true, areg[0], true, breg[nt][0],
                                                     acc, false, false);
        acc = __builtin_amdgcn_wmma_i32_16x16x64_iu8(true, areg[1], true, breg[nt][1],
                                                     acc, false, false);
        float* op = out_base + nt * 16;
#pragma unroll
        for (int r = 0; r < 8; ++r) {
            __builtin_nontemporal_store((float)acc[r] * alpha, op + (size_t)r * kN);
        }
    }
}

extern "C" void kernel_launch(void* const* d_in, const int* in_sizes, int n_in,
                              void* d_out, int out_size, void* d_ws, size_t ws_size,
                              hipStream_t stream) {
    const int8_t* a     = (const int8_t*)d_in[0];
    const int8_t* b     = (const int8_t*)d_in[1];
    const float*  alpha = (const float*)d_in[2];
    float*        out   = (float*)d_out;

    const int total_waves = kB * (kM / 16) * (kN / 64);
    const int blocks      = total_waves / kWavesPerBlock;
    bmm_s8t_s8n_f32t_wmma<<<blocks, 256, 0, stream>>>(a, b, alpha, out);
}
